// RNN_76407468196225
// MI455X (gfx1250) — matrix operand order repaired and checked by reading, not compiled, not run on hardware
//
#include <hip/hip_runtime.h>
#include <hip/hip_bf16.h>

// ---------------------------------------------------------------------------
// 2-layer ReLU RNN for MI455X (gfx1250), WMMA f32<-f16 path.
//
// Plan:
//   prep:    fp32->fp16 weight conversion, bias fusion (bih+bhh)
//   xp0:     xp0[t][b][j] = dot8(normalize(X[b][t][:]), Wih0[j][:]) + b0[j]
//   rnn L0:  persistent 128-WG kernel; 8 batch groups x 16 col tiles.
//            Whh0 64x1024 f16 slice cached in LDS (128KB) for all 512 steps.
//            Per step: 32x v_wmma_f32_16x16x32_f16 per wave, relu, store y0,
//            per-group inter-WG barrier via monotonic atomic counter.
//   xp1:     parallel WMMA GEMM  xp = y0 @ Wih1^T + b1   (137 GFLOP, off the
//            serial critical path)
//   rnn L1:  same persistent kernel, ping-pong h buffers, keep final h only
//   fc:      out = h_final @ Wfc^T + bfc   (tiny)
// ---------------------------------------------------------------------------

typedef __attribute__((ext_vector_type(16))) _Float16 v16h;
typedef __attribute__((ext_vector_type(8)))  _Float16 v8h;
typedef __attribute__((ext_vector_type(8)))  float    v8f;

#define Bdim 128
#define Tdim 512
#define Idim 8
#define Hdim 1024
#define Odim 4

// ---- WMMA fragment loaders (ISA 7.12.2 layouts, row-major source) ----------
// A 16x32 f16: lane l holds row m=l&15; halves 0..7 -> K = kh*8+0..7,
// halves 8..15 -> K = 16+kh*8+0..7, kh = l>>4.  Two 16B contiguous loads.
template <typename P>
static __device__ __forceinline__ v16h load_fragA(P rowptr, int k0, int kh) {
  union { v16h v; v8h h[2]; } u;
  u.h[0] = *(const v8h*)(rowptr + k0 + kh * 8);
  u.h[1] = *(const v8h*)(rowptr + k0 + 16 + kh * 8);
  return u.v;
}
// B 32x16 f16 (K x N), B[k][n] = W[n][k]: lane l holds col n=l&15,
// halves 0..7 -> K = kh*8+0..7, halves 8..15 -> K = 16+kh*8+0..7 (same as A).
template <typename P>
static __device__ __forceinline__ v16h load_fragB(P colptr, int k0, int kh) {
  union { v16h v; v8h h[2]; } u;
  u.h[0] = *(const v8h*)(colptr + k0 + kh * 8);
  u.h[1] = *(const v8h*)(colptr + k0 + 16 + kh * 8);
  return u.v;
}

// ---------------------------------------------------------------------------
__global__ void prep_kernel(const float* __restrict__ Whh0,
                            const float* __restrict__ Whh1,
                            const float* __restrict__ Wih1,
                            const float* __restrict__ bih0, const float* __restrict__ bhh0,
                            const float* __restrict__ bih1, const float* __restrict__ bhh1,
                            _Float16* __restrict__ WhhA, _Float16* __restrict__ WhhB,
                            _Float16* __restrict__ W1f, float* __restrict__ b0,
                            float* __restrict__ b1) {
  size_t i = (size_t)blockIdx.x * blockDim.x + threadIdx.x;
  if (i < (size_t)Hdim * Hdim) {
    WhhA[i] = (_Float16)Whh0[i];
    WhhB[i] = (_Float16)Whh1[i];
    W1f[i]  = (_Float16)Wih1[i];
  }
  if (i < Hdim) {
    b0[i] = bih0[i] + bhh0[i];
    b1[i] = bih1[i] + bhh1[i];
  }
}

// xp0[t][b][j] : K=8 input projection with RPM normalization. HBM-bound, tiny.
__global__ void xp0_kernel(const float* __restrict__ X,     // [B][T][I]
                           const float* __restrict__ Wih0,  // [H][I]
                           const float* __restrict__ b0,    // [H]
                           _Float16* __restrict__ XP) {     // [T][B][H]
  size_t idx = (size_t)blockIdx.x * blockDim.x + threadIdx.x;
  int j = (int)(idx & (Hdim - 1));
  size_t tb = idx >> 10;
  int b = (int)(tb & (Bdim - 1));
  int t = (int)(tb >> 7);
  if (t >= Tdim) return;
  const float* xr = X + ((size_t)b * Tdim + t) * Idim;
  const float* wr = Wih0 + (size_t)j * Idim;
  float acc = b0[j];
#pragma unroll
  for (int i = 0; i < Idim; ++i)
    acc += ((xr[i] - 1500.0f) * (1.0f / 7500.0f)) * wr[i];
  XP[idx] = (_Float16)acc;
}

// Parallel WMMA GEMM: XP[r][n] = sum_k Y[r][k] * W[n][k] + bias[n]
// R = T*B = 65536 rows. One 16x16 tile per wave, K loop of 32 WMMAs.
__global__ void __launch_bounds__(256)
xp_gemm_kernel(const _Float16* __restrict__ Y,   // [R][H] f16
               const _Float16* __restrict__ W,   // [H][H] f16 (Wih1)
               const float* __restrict__ bias,   // [H]
               _Float16* __restrict__ XP) {      // [R][H] f16
  const int wave = threadIdx.x >> 5;
  const int lane = threadIdx.x & 31;
  const int gw = blockIdx.x * 8 + wave;
  const int rowTile = gw >> 6;            // 4096 row tiles
  const int colTile = gw & 63;            // 64 col tiles
  const int r0 = rowTile * 16;
  const int n0 = colTile * 16;
  const int mn = lane & 15, kh = lane >> 4;
  const _Float16* arow = Y + (size_t)(r0 + mn) * Hdim;
  const _Float16* bcol = W + (size_t)(n0 + mn) * Hdim;
  v8f c = {};
#pragma unroll 4
  for (int k0 = 0; k0 < Hdim; k0 += 32) {
    v16h a = load_fragA(arow, k0, kh);
    v16h b = load_fragB(bcol, k0, kh);
    c = __builtin_amdgcn_wmma_f32_16x16x32_f16(false, a, false, b, (short)0, c,
                                               false, false);
  }
  const float bb = bias[n0 + mn];         // n = lane&15
#pragma unroll
  for (int r = 0; r < 8; ++r) {
    int mm = r + kh * 8;                  // C layout: M = r + 8*(lane>>4)
    XP[(size_t)(r0 + mm) * Hdim + n0 + mn] = (_Float16)(c[r] + bb);
  }
}

// Persistent recurrent kernel. 128 WGs of 128 threads (4 waves).
// blockIdx = g*16 + jt : batch group g (16 rows), column tile jt (64 cols).
// Whh slice [64 x 1024] f16 cached in LDS for the whole time loop.
__global__ void __launch_bounds__(128, 1)
rnn_layer_kernel(const _Float16* __restrict__ XP,   // [T][B][H] precomputed proj
                 const _Float16* __restrict__ Whh,  // [H][H] f16
                 _Float16* __restrict__ ybuf,       // [T][B][H] (store_all=1)
                 _Float16* __restrict__ hp,         // [2][B][H] (store_all=0)
                 unsigned* __restrict__ ctr,        // [8] per-group counters
                 int store_all) {
  extern __shared__ _Float16 sW[];                  // [64][H] = 128 KB
  const int tid = threadIdx.x;
  const int wave = tid >> 5, lane = tid & 31;
  const int g  = blockIdx.x >> 4;                   // batch group 0..7
  const int jt = blockIdx.x & 15;                   // col tile 0..15
  const int b0 = g * 16;
  const int j0 = jt * 64;

  // one-time LDS fill of this WG's Whh slice (rows j0..j0+63)
  for (int i = tid; i < 64 * (Hdim / 8); i += 128) {
    int row = i >> 7;                               // Hdim/8 = 128 chunks/row
    int c8  = i & 127;
    *(v8h*)&sW[row * Hdim + c8 * 8] =
        *(const v8h*)&Whh[(size_t)(j0 + row) * Hdim + c8 * 8];
  }
  __syncthreads();

  const int mn = lane & 15, kh = lane >> 4;
  const int nloc = wave * 16 + mn;                  // local col within 64
  const int jcol = j0 + nloc;
  unsigned* myctr = ctr + g;

  for (int t = 0; t < Tdim; ++t) {
    v8f c = {};
    if (t > 0) {
      const _Float16* hin =
          store_all ? (ybuf + ((size_t)(t - 1) * Bdim + b0 + mn) * Hdim)
                    : (hp + ((size_t)((t + 1) & 1) * Bdim + b0 + mn) * Hdim);
      const _Float16* brow = &sW[(size_t)nloc * Hdim];  // LDS, ds_load_b128
#pragma unroll 4
      for (int k0 = 0; k0 < Hdim; k0 += 32) {
        v16h a = load_fragA(hin, k0, kh);
        v16h b = load_fragB(brow, k0, kh);
        c = __builtin_amdgcn_wmma_f32_16x16x32_f16(false, a, false, b,
                                                   (short)0, c, false, false);
      }
    }
    // h_new = relu(c + xp[t]); write y0 / ping-pong h
    const _Float16* xpt = XP + (size_t)t * Bdim * Hdim;
    _Float16* hout = store_all ? (ybuf + (size_t)t * Bdim * Hdim)
                               : (hp + (size_t)(t & 1) * Bdim * Hdim);
#pragma unroll
    for (int r = 0; r < 8; ++r) {
      int mm = r + kh * 8;
      size_t off = (size_t)(b0 + mm) * Hdim + jcol;
      float v = c[r] + (float)xpt[off];
      v = v > 0.0f ? v : 0.0f;
      hout[off] = (_Float16)v;
    }
    // prefetch next step's xp tile (global_prefetch_b8)
    if (t + 1 < Tdim)
      __builtin_prefetch(XP + ((size_t)(t + 1) * Bdim + b0 + mn) * Hdim + j0, 0, 1);

    // per-group inter-WG barrier: monotonic counter, target 16*(t+1)
    __threadfence();
    __syncthreads();
    if (tid == 0) {
      atomicAdd(myctr, 1u);
      unsigned target = 16u * (unsigned)(t + 1);
      while (__atomic_load_n(myctr, __ATOMIC_ACQUIRE) < target)
        __builtin_amdgcn_s_sleep(1);
    }
    __syncthreads();
    __threadfence();
  }
}

__global__ void fc_kernel(const _Float16* __restrict__ h,   // [B][H] final
                          const float* __restrict__ Wfc,    // [O][H]
                          const float* __restrict__ bfc,    // [O]
                          float* __restrict__ out) {        // [B][O]
  int idx = blockIdx.x * blockDim.x + threadIdx.x;
  if (idx >= Bdim * Odim) return;
  int o = idx & (Odim - 1);
  int b = idx >> 2;
  const _Float16* hr = h + (size_t)b * Hdim;
  const float* wr = Wfc + (size_t)o * Hdim;
  float acc = bfc[o];
  for (int k = 0; k < Hdim; ++k) acc += (float)hr[k] * wr[k];
  out[idx] = acc;
}

// ---------------------------------------------------------------------------
extern "C" void kernel_launch(void* const* d_in, const int* in_sizes, int n_in,
                              void* d_out, int out_size, void* d_ws, size_t ws_size,
                              hipStream_t stream) {
  const float* X    = (const float*)d_in[0];
  const float* Wih0 = (const float*)d_in[1];
  const float* Whh0 = (const float*)d_in[2];
  const float* bih0 = (const float*)d_in[3];
  const float* bhh0 = (const float*)d_in[4];
  const float* Wih1 = (const float*)d_in[5];
  const float* Whh1 = (const float*)d_in[6];
  const float* bih1 = (const float*)d_in[7];
  const float* bhh1 = (const float*)d_in[8];
  const float* Wfc  = (const float*)d_in[9];
  const float* bfc  = (const float*)d_in[10];

  char* ws = (char*)d_ws;
  const size_t szHH = (size_t)Hdim * Hdim * sizeof(_Float16);   // 2 MB
  const size_t szTBH = (size_t)Tdim * Bdim * Hdim * sizeof(_Float16); // 128 MB
  _Float16* WhhA = (_Float16*)(ws);
  _Float16* WhhB = (_Float16*)(ws + szHH);
  _Float16* W1f  = (_Float16*)(ws + 2 * szHH);
  float*    b0   = (float*)(ws + 3 * szHH);
  float*    b1   = (float*)(ws + 3 * szHH + 4096);
  unsigned* ctrs = (unsigned*)(ws + 3 * szHH + 8192);           // 512 B (2x8 used)
  _Float16* hp   = (_Float16*)(ws + 3 * szHH + 8704);           // 512 KB
  _Float16* XP   = (_Float16*)(ws + 3 * szHH + 8704 + 2 * (size_t)Bdim * Hdim * 2);
  _Float16* Y0   = (_Float16*)((char*)XP + szTBH);
  // total ws use ~= 263 MB

  hipMemsetAsync(ctrs, 0, 512, stream);

  prep_kernel<<<(Hdim * Hdim + 255) / 256, 256, 0, stream>>>(
      Whh0, Whh1, Wih1, bih0, bhh0, bih1, bhh1, WhhA, WhhB, W1f, b0, b1);

  xp0_kernel<<<(Tdim * Bdim * Hdim) / 256, 256, 0, stream>>>(X, Wih0, b0, XP);

  // layer 0: read/write h through Y0 itself (h at step t lives at Y0[t])
  rnn_layer_kernel<<<128, 128, 64 * Hdim * sizeof(_Float16), stream>>>(
      XP, WhhA, Y0, hp, ctrs, /*store_all=*/1);

  // layer 1 input projection off the critical path: XP = Y0 @ Wih1^T + b1
  xp_gemm_kernel<<<(Tdim * Bdim / 16) * (Hdim / 16) / 8, 256, 0, stream>>>(
      Y0, W1f, b1, XP);

  // layer 1 recurrence: ping-pong h, final h ends in hp[(T-1)&1] = hp[1]
  rnn_layer_kernel<<<128, 128, 64 * Hdim * sizeof(_Float16), stream>>>(
      XP, WhhB, Y0, hp, ctrs + 8, /*store_all=*/0);

  fc_kernel<<<(Bdim * Odim + 255) / 256, 256, 0, stream>>>(
      hp + (size_t)Bdim * Hdim, Wfc, bfc, (float*)d_out);
}
